// Auto_Attn_73735998538008
// MI455X (gfx1250) — compile-verified, matrix operand order checked by reading, not run on hardware
//
#include <hip/hip_runtime.h>
#include <hip/hip_bf16.h>
#include <stdint.h>

#define TPB 256
#define BATCH 2
#define CH 256
#define CH2 512
#define NPIX 4096
#define CQ 64
#define PW 66   // 64 + 1 halo each side

typedef __attribute__((ext_vector_type(16))) __bf16 v16bf;
typedef __attribute__((ext_vector_type(8)))  float  v8f;

#if __has_builtin(__builtin_amdgcn_global_load_async_to_lds_b128)
#define HAVE_ASYNC_LDS 1
#endif

// ---------- bf16 helpers (round-to-nearest-even) ----------
static __device__ __forceinline__ unsigned short f2bf(float f) {
  union { float f; uint32_t u; } c; c.f = f;
  uint32_t x = c.u;
  uint32_t r = x + 0x7FFFu + ((x >> 16) & 1u);
  return (unsigned short)(r >> 16);
}
static __device__ __forceinline__ float bf2f(unsigned short u) {
  union { uint32_t u; float f; } c; c.u = ((uint32_t)u) << 16; return c.f;
}

// ---------- WMMA wave-tile primitives ----------
union TileU { v16bf v; uint4 q[2]; };

// Load a 16x32 bf16 tile in the documented A/B register layout.
// lane<16:  halves 0..7 = K[0..7],  halves 8..15 = K[16..23]   (row/col = lane)
// lane>=16: halves 0..7 = K[8..15], halves 8..15 = K[24..31]   (row/col = lane-16)
static __device__ __forceinline__ v16bf load_tile(const unsigned short* base, int ldk, int lane) {
  int rc = lane & 15;
  int hi = (lane >> 4) << 3;            // 0 or 8
  const unsigned short* p = base + rc * ldk + hi;
  TileU t;
  t.q[0] = *(const uint4*)(p);
  t.q[1] = *(const uint4*)(p + 16);
  return t.v;
}

static __device__ __forceinline__ v8f vzero() {
  v8f z;
  #pragma unroll
  for (int r = 0; r < 8; ++r) z[r] = 0.0f;
  return z;
}

static __device__ __forceinline__ v8f wmma_bf16(v16bf a, v16bf b, v8f c) {
  return __builtin_amdgcn_wmma_f32_16x16x32_bf16(false, a, false, b, (short)0, c, false, false);
}

// C/D layout: VGPR r -> row = r + 8*(lane>=16), col = lane&15
static __device__ __forceinline__ void store_f32(float* base, int ldn, int lane, v8f c) {
  int col = lane & 15;
  int ro = (lane >> 4) << 3;
  #pragma unroll
  for (int r = 0; r < 8; ++r) base[(size_t)(r + ro) * ldn + col] = c[r];
}
static __device__ __forceinline__ void store_bf16(unsigned short* base, int ldn, int lane, v8f c) {
  int col = lane & 15;
  int ro = (lane >> 4) << 3;
  #pragma unroll
  for (int r = 0; r < 8; ++r) base[(size_t)(r + ro) * ldn + col] = f2bf(c[r]);
}

// ---------- global -> LDS staging of one 16B chunk ----------
static __device__ __forceinline__ void stage_b128(const unsigned short* g, unsigned short* l) {
#if defined(HAVE_ASYNC_LDS)
  // Builtin signature (probed via diagnostic): (v4i32 AS1*, v4i32 AS3*, imm offset, imm cpol).
  // Generic->AS casts go through integers; low 32 bits of a generic shared pointer are the
  // LDS offset on amdgcn, so the u32 truncation for the AS3 pointer is exact.
  typedef int v4i_vs __attribute__((vector_size(16)));
  typedef __attribute__((address_space(1))) v4i_vs* gp_t;
  typedef __attribute__((address_space(3))) v4i_vs* lp_t;
  __builtin_amdgcn_global_load_async_to_lds_b128(
      (gp_t)(unsigned long long)(uintptr_t)g,
      (lp_t)(unsigned int)(uintptr_t)l, 0, 0);
#else
  *(uint4*)l = *(const uint4*)g;
#endif
}
static __device__ __forceinline__ void stage_fence() {
#if defined(HAVE_ASYNC_LDS)
#if __has_builtin(__builtin_amdgcn_s_wait_asynccnt)
  __builtin_amdgcn_s_wait_asynccnt(0);
#else
  asm volatile("s_wait_asynccnt 0" ::: "memory");
#endif
#endif
  __syncthreads();
}

// ---------- block reduction ----------
static __device__ __forceinline__ float block_reduce_sum(float v, float* sbuf) {
  int t = threadIdx.x;
  sbuf[t] = v; __syncthreads();
  for (int s = TPB >> 1; s > 0; s >>= 1) {
    if (t < s) sbuf[t] += sbuf[t + s];
    __syncthreads();
  }
  float r = sbuf[0]; __syncthreads();
  return r;
}

// ================= conversion kernels =================
__global__ void k_convert(const float* __restrict__ x, const float* __restrict__ pre,
                          unsigned short* xb_cn, unsigned short* xbT, unsigned short* preb_cn) {
  int i = blockIdx.x * TPB + threadIdx.x;           // over BATCH*CH*NPIX
  int n  = i & (NPIX - 1);
  int bc = i >> 12;
  int c  = bc & (CH - 1);
  int b  = bc >> 8;
  float xv = x[i];
  xb_cn[i] = f2bf(xv);
  xbT[((size_t)b * NPIX + n) * CH + c] = f2bf(xv);
  preb_cn[i] = f2bf(pre[i]);
}

__global__ void k_convert_wq(const float* __restrict__ qw, unsigned short* wqb) {
  int i = blockIdx.x * TPB + threadIdx.x;           // CQ*CH
  wqb[i] = f2bf(qw[i]);
}

__global__ void k_zero_u16(unsigned short* p, size_t n) {
  size_t i = (size_t)blockIdx.x * TPB + threadIdx.x;
  size_t st = (size_t)gridDim.x * TPB;
  for (; i < n; i += st) p[i] = 0;
}

// ================= q = x^T * Wq^T : qT[b][n][cq] =================
__global__ void k_qgemm(const unsigned short* __restrict__ xbT,
                        const unsigned short* __restrict__ wqb,
                        unsigned short* qT) {
  int tile = blockIdx.x * 8 + (threadIdx.x >> 5);   // 2048 tiles: b*1024 + mt*4 + nt
  int lane = threadIdx.x & 31;
  int nt = tile & 3;
  int mt = (tile >> 2) & 255;
  int b  = tile >> 10;
  const unsigned short* ab = xbT + ((size_t)b * NPIX + mt * 16) * CH;
  const unsigned short* bb = wqb + (size_t)(nt * 16) * CH;
  v8f acc = vzero();
  #pragma unroll 2
  for (int k = 0; k < CH; k += 32)
    acc = wmma_bf16(load_tile(ab + k, CH, lane), load_tile(bb + k, CH, lane), acc);
  store_bf16(qT + ((size_t)b * NPIX + mt * 16) * CQ + nt * 16, CQ, lane, acc);
}

// ===== energy[b][i][j] = sum_cq qT[i][cq]*qT[j][cq], 4-wide j blocking (A reuse) =====
__global__ void k_energy(const unsigned short* __restrict__ qT, float* E) {
  int tile = blockIdx.x * 8 + (threadIdx.x >> 5);   // 32768 tiles: b*16384 + it*64 + jb
  int lane = threadIdx.x & 31;
  int jb = tile & 63;
  int it = (tile >> 6) & 255;
  int b  = tile >> 14;
  const unsigned short* qb = qT + (size_t)b * NPIX * CQ;
  const unsigned short* ap = qb + (size_t)(it * 16) * CQ;
  v16bf a0 = load_tile(ap,      CQ, lane);
  v16bf a1 = load_tile(ap + 32, CQ, lane);
  #pragma unroll
  for (int j4 = 0; j4 < 4; ++j4) {
    const unsigned short* bp = qb + (size_t)(jb * 64 + j4 * 16) * CQ;
    v8f acc = vzero();
    acc = wmma_bf16(a0, load_tile(bp,      CQ, lane), acc);
    acc = wmma_bf16(a1, load_tile(bp + 32, CQ, lane), acc);
    store_f32(E + (size_t)b * NPIX * NPIX + (size_t)(it * 16) * NPIX + jb * 64 + j4 * 16,
              NPIX, lane, acc);
  }
}

// ================= row softmax (fp32 in place -> d_out) + bf16 copy =================
__global__ void k_softmax(float* E, unsigned short* attnb) {
  __shared__ float sbuf[TPB];
  int row = blockIdx.x;                              // 0..8191
  float* r = E + (size_t)row * NPIX;
  unsigned short* rb = attnb + (size_t)row * NPIX;
  int t = threadIdx.x;
  float m = -3.4e38f;
  for (int j = t; j < NPIX; j += TPB) m = fmaxf(m, r[j]);
  sbuf[t] = m; __syncthreads();
  for (int s = TPB >> 1; s > 0; s >>= 1) { if (t < s) sbuf[t] = fmaxf(sbuf[t], sbuf[t + s]); __syncthreads(); }
  m = sbuf[0]; __syncthreads();
  float sum = 0.f;
  for (int j = t; j < NPIX; j += TPB) { float e = __expf(r[j] - m); r[j] = e; sum += e; }
  sum = block_reduce_sum(sum, sbuf);
  float inv = 1.0f / sum;
  for (int j = t; j < NPIX; j += TPB) { float v = r[j] * inv; r[j] = v; rb[j] = f2bf(v); }
}

// ===== outT[b][j][c] = sum_k attn[j][k]*x[c][k]; cfT same with pre =====
// Workgroup-cooperative: block stages A(64x64), Bx(64x64), Bp(64x64) into LDS
// (async-to-LDS when available); 8 waves: wave = {matrix m, local jt}, 4 ct accums each.
#define SLD 72   // LDS row stride in halves (16B aligned, staggers banks)
__global__ void k_attn_apply(const unsigned short* __restrict__ attnb,
                             const unsigned short* __restrict__ xb_cn,
                             const unsigned short* __restrict__ preb_cn,
                             float* outT, float* cfT) {
  __shared__ __align__(16) unsigned short sA[64 * SLD];
  __shared__ __align__(16) unsigned short sBx[64 * SLD];
  __shared__ __align__(16) unsigned short sBp[64 * SLD];
  int blk = blockIdx.x;           // 512 blocks: b*256 + jb*4 + cb
  int cb = blk & 3;
  int jb = (blk >> 2) & 63;
  int b  = blk >> 8;
  int tid  = threadIdx.x;
  int wave = tid >> 5, lane = tid & 31;
  int m  = wave >> 2;             // 0 -> out (x), 1 -> cf (pre)
  int jt = wave & 3;              // local j tile within the 64-row block
  const unsigned short* Ag  = attnb   + ((size_t)b * NPIX + jb * 64) * (size_t)NPIX;
  const unsigned short* Bxg = xb_cn   + ((size_t)b * CH   + cb * 64) * (size_t)NPIX;
  const unsigned short* Bpg = preb_cn + ((size_t)b * CH   + cb * 64) * (size_t)NPIX;
  v8f acc[4];
  #pragma unroll
  for (int c4 = 0; c4 < 4; ++c4) acc[c4] = vzero();

  for (int k0 = 0; k0 < NPIX; k0 += 64) {
    // stage 3 x (64 rows x 64 halves) = 3 x 512 b128 chunks, 256 threads -> 2 each
    for (int e = tid; e < 512; e += TPB) {
      int r = e >> 3, cc = (e & 7) << 3;          // row, half-offset within chunk row
      size_t go = (size_t)r * NPIX + k0 + cc;
      int    lo = r * SLD + cc;
      stage_b128(Ag  + go, sA  + lo);
      stage_b128(Bxg + go, sBx + lo);
      stage_b128(Bpg + go, sBp + lo);
      __builtin_prefetch(Ag + go + 64, 0, 0);     // next chunk hint
    }
    stage_fence();
    const unsigned short* Bs = (m == 0) ? sBx : sBp;
    #pragma unroll
    for (int kk = 0; kk < 64; kk += 32) {
      v16bf a = load_tile(sA + jt * 16 * SLD + kk, SLD, lane);
      #pragma unroll
      for (int c4 = 0; c4 < 4; ++c4)
        acc[c4] = wmma_bf16(a, load_tile(Bs + c4 * 16 * SLD + kk, SLD, lane), acc[c4]);
    }
    __syncthreads();
  }
  float* dst = (m == 0 ? outT : cfT) + ((size_t)b * NPIX + jb * 64 + jt * 16) * CH + cb * 64;
  #pragma unroll
  for (int c4 = 0; c4 < 4; ++c4) store_f32(dst + c4 * 16, CH, lane, acc[c4]);
}

// ========== build y = concat(gamma*out + x, alpha*(1-m)*cf + m*pre), NHWC padded bf16 ==========
__global__ void k_build_y(const float* __restrict__ outT, const float* __restrict__ cfT,
                          const float* __restrict__ x, const float* __restrict__ pre,
                          const float* __restrict__ mask, const float* __restrict__ gamma_p,
                          const float* __restrict__ alpha_p, unsigned short* ybp) {
  int i = blockIdx.x * TPB + threadIdx.x;            // over BATCH*NPIX*CH2
  int ch = i & (CH2 - 1);
  int bn = i >> 9;
  int n  = bn & (NPIX - 1);
  int b  = bn >> 12;
  float v;
  if (ch < CH) {
    float o  = outT[((size_t)b * NPIX + n) * CH + ch];
    float xv = x[((size_t)b * CH + ch) * NPIX + n];
    v = gamma_p[0] * o + xv;
  } else {
    int c2 = ch - CH;
    float cf = cfT[((size_t)b * NPIX + n) * CH + c2];
    float mk = mask[(size_t)b * NPIX + n];
    float pv = pre[((size_t)b * CH + c2) * NPIX + n];
    v = alpha_p[0] * (1.0f - mk) * cf + mk * pv;
  }
  int y = n >> 6, xp = n & 63;
  ybp[(((size_t)b * PW + (y + 1)) * PW + (xp + 1)) * CH2 + ch] = f2bf(v);
}

// ================= BN statistics =================
__global__ void k_bnstats_pad(const unsigned short* __restrict__ src, int Cc,
                              const float* __restrict__ scale, const float* __restrict__ bias,
                              float* acoef, float* bcoef) {
  __shared__ float s1[TPB]; __shared__ float s2[TPB];
  int ch = blockIdx.x, t = threadIdx.x;
  float a = 0.f, q = 0.f;
  for (int e = t; e < BATCH * NPIX; e += TPB) {
    int b = e >> 12, n = e & (NPIX - 1);
    int y = n >> 6, xp = n & 63;
    float v = bf2f(src[(((size_t)b * PW + (y + 1)) * PW + (xp + 1)) * Cc + ch]);
    a += v; q += v * v;
  }
  s1[t] = a; s2[t] = q; __syncthreads();
  for (int s = TPB >> 1; s > 0; s >>= 1) { if (t < s) { s1[t] += s1[t + s]; s2[t] += s2[t + s]; } __syncthreads(); }
  if (t == 0) {
    float inv_n = 1.0f / (BATCH * NPIX);
    float mean = s1[0] * inv_n;
    float var  = s2[0] * inv_n - mean * mean;
    float aa = scale[ch] * rsqrtf(var + 1e-5f);
    acoef[ch] = aa; bcoef[ch] = bias[ch] - mean * aa;
  }
}

__global__ void k_bnstats_f32(const float* __restrict__ src, int Cc,
                              const float* __restrict__ scale, const float* __restrict__ bias,
                              float* acoef, float* bcoef) {
  __shared__ float s1[TPB]; __shared__ float s2[TPB];
  int ch = blockIdx.x, t = threadIdx.x;
  float a = 0.f, q = 0.f;
  for (int e = t; e < BATCH * NPIX; e += TPB) {
    float v = src[(size_t)e * Cc + ch];
    a += v; q += v * v;
  }
  s1[t] = a; s2[t] = q; __syncthreads();
  for (int s = TPB >> 1; s > 0; s >>= 1) { if (t < s) { s1[t] += s1[t + s]; s2[t] += s2[t + s]; } __syncthreads(); }
  if (t == 0) {
    float inv_n = 1.0f / (BATCH * NPIX);
    float mean = s1[0] * inv_n;
    float var  = s2[0] * inv_n - mean * mean;
    float aa = scale[ch] * rsqrtf(var + 1e-5f);
    acoef[ch] = aa; bcoef[ch] = bias[ch] - mean * aa;
  }
}

// ================= BN apply + LeakyReLU =================
__global__ void k_act_pad(const unsigned short* __restrict__ src, unsigned short* dst,
                          const float* __restrict__ A, const float* __restrict__ Bc, int cshift) {
  int i = blockIdx.x * TPB + threadIdx.x;
  int Cc = 1 << cshift;
  int ch = i & (Cc - 1);
  int bn = i >> cshift;
  int n  = bn & (NPIX - 1);
  int b  = bn >> 12;
  int y = n >> 6, xp = n & 63;
  size_t idx = (((size_t)b * PW + (y + 1)) * PW + (xp + 1)) * Cc + ch;
  float v = A[ch] * bf2f(src[idx]) + Bc[ch];
  v = v > 0.f ? v : 0.01f * v;
  dst[idx] = f2bf(v);
}

__global__ void k_act_f32(const float* __restrict__ src, unsigned short* dst,
                          const float* __restrict__ A, const float* __restrict__ Bc) {
  int i = blockIdx.x * TPB + threadIdx.x;            // over BATCH*NPIX*CH, src is [b][n][c]
  int ch = i & (CH - 1);
  int bn = i >> 8;
  int n  = bn & (NPIX - 1);
  int b  = bn >> 12;
  float v = A[ch] * src[i] + Bc[ch];
  v = v > 0.f ? v : 0.01f * v;
  int y = n >> 6, xp = n & 63;
  dst[(((size_t)b * PW + (y + 1)) * PW + (xp + 1)) * CH + ch] = f2bf(v);
}

// ===== conv3x3 implicit GEMM, 4-wide cout blocking (A reuse), Cout=256 =====
__global__ void k_conv3x3(const unsigned short* __restrict__ src, const unsigned short* __restrict__ wt,
                          float* dst, int Cin) {
  int tile = blockIdx.x * 8 + (threadIdx.x >> 5);    // 2048 tiles: (b*256 + mt)*4 + ntb
  int lane = threadIdx.x & 31;
  int ntb = tile & 3;
  int rest = tile >> 2;
  int mt = rest & 255;
  int b  = rest >> 8;
  int y = mt >> 2, x0 = (mt & 3) * 16;
  v8f acc[4];
  #pragma unroll
  for (int c4 = 0; c4 < 4; ++c4) acc[c4] = vzero();
  for (int t = 0; t < 9; ++t) {
    int dy = t / 3 - 1, dx = t % 3 - 1;
    const unsigned short* ab = src + (((size_t)b * PW + (y + 1 + dy)) * PW + (x0 + 1 + dx)) * Cin;
    const unsigned short* bb = wt + ((size_t)(t * CH + ntb * 64)) * Cin;
    for (int k = 0; k < Cin; k += 32) {
      v16bf a = load_tile(ab + k, Cin, lane);
      #pragma unroll
      for (int c4 = 0; c4 < 4; ++c4)
        acc[c4] = wmma_bf16(a, load_tile(bb + (size_t)c4 * 16 * Cin + k, Cin, lane), acc[c4]);
    }
  }
  float* db = dst + ((size_t)b * NPIX + (y << 6) + x0) * CH + ntb * 64;
  #pragma unroll
  for (int c4 = 0; c4 < 4; ++c4) store_f32(db + c4 * 16, CH, lane, acc[c4]);
}

// ===== conv1x1 shortcut (512 -> 256), 4-wide cout blocking =====
__global__ void k_conv1x1(const unsigned short* __restrict__ ybp, const unsigned short* __restrict__ byt,
                          float* dst) {
  int tile = blockIdx.x * 8 + (threadIdx.x >> 5);    // 2048 tiles
  int lane = threadIdx.x & 31;
  int ntb = tile & 3;
  int rest = tile >> 2;
  int mt = rest & 255;
  int b  = rest >> 8;
  int y = mt >> 2, x0 = (mt & 3) * 16;
  const unsigned short* ab = ybp + (((size_t)b * PW + (y + 1)) * PW + (x0 + 1)) * CH2;
  const unsigned short* bb = byt + (size_t)(ntb * 64) * CH2;
  v8f acc[4];
  #pragma unroll
  for (int c4 = 0; c4 < 4; ++c4) acc[c4] = vzero();
  for (int k = 0; k < CH2; k += 32) {
    v16bf a = load_tile(ab + k, CH2, lane);
    #pragma unroll
    for (int c4 = 0; c4 < 4; ++c4)
      acc[c4] = wmma_bf16(a, load_tile(bb + (size_t)c4 * 16 * CH2 + k, CH2, lane), acc[c4]);
  }
  float* db = dst + ((size_t)b * NPIX + (y << 6) + x0) * CH + ntb * 64;
  #pragma unroll
  for (int c4 = 0; c4 < 4; ++c4) store_f32(db + c4 * 16, CH, lane, acc[c4]);
}

// ================= res = h2 + sc, back to NCHW =================
__global__ void k_final(const float* __restrict__ h2f, const float* __restrict__ scf, float* res) {
  int i = blockIdx.x * TPB + threadIdx.x;            // [b][n][c] order
  int c  = i & (CH - 1);
  int bn = i >> 8;
  int n  = bn & (NPIX - 1);
  int b  = bn >> 12;
  res[((size_t)b * CH + c) * NPIX + n] = h2f[i] + scf[i];
}

// ================= spectral norm: one power iteration, h = 256 =================
__global__ void k_sn(const float* __restrict__ w, const float* __restrict__ u, int wd,
                     float* vtmp, float* utmp, float* inv_sigma) {
  __shared__ float sbuf[TPB];
  int t = threadIdx.x;
  float ss = 0.f;                                    // v_unnorm = W^T u
  for (int j = t; j < wd; j += TPB) {
    float acc = 0.f;
    for (int i2 = 0; i2 < CH; ++i2) acc += w[(size_t)i2 * wd + j] * u[i2];
    vtmp[j] = acc; ss += acc * acc;
  }
  ss = block_reduce_sum(ss, sbuf);
  float sv = 1.0f / (sqrtf(ss) + 1e-12f);            // v = vtmp * sv
  float ss2 = 0.f;                                   // u_unnorm = W v
  for (int i2 = t; i2 < CH; i2 += TPB) {
    float acc = 0.f;
    for (int j = 0; j < wd; ++j) acc += w[(size_t)i2 * wd + j] * vtmp[j];
    acc *= sv;
    utmp[i2] = acc; ss2 += acc * acc;
  }
  ss2 = block_reduce_sum(ss2, sbuf);
  if (t == 0) {
    float nu = sqrtf(ss2);
    float sigma = ss2 / (nu + 1e-12f);               // u' . (W v)
    inv_sigma[0] = 1.0f / sigma;
  }
}

// transform: wt[ky][kx][co][ci] = bf16(w[co][ci][ky][kx] / sigma)
__global__ void k_wt3x3(const float* __restrict__ w, unsigned short* wt,
                        const float* __restrict__ inv_sigma, int Cin) {
  int i = blockIdx.x * TPB + threadIdx.x;            // exactly CH*Cin*9 threads
  int kx = i % 3; int r = i / 3;
  int ky = r % 3; r /= 3;
  int ci = r % Cin; int co = r / Cin;
  float v = w[i] * inv_sigma[0];
  wt[(((size_t)(ky * 3 + kx)) * CH + co) * Cin + ci] = f2bf(v);
}

__global__ void k_wt1x1(const float* __restrict__ w, unsigned short* wt,
                        const float* __restrict__ inv_sigma) {
  int i = blockIdx.x * TPB + threadIdx.x;            // CH*CH2, layout already [co][ci]
  wt[i] = f2bf(w[i] * inv_sigma[0]);
}

// ========================================================================
extern "C" void kernel_launch(void* const* d_in, const int* in_sizes, int n_in,
                              void* d_out, int out_size, void* d_ws, size_t ws_size,
                              hipStream_t stream) {
  (void)in_sizes; (void)n_in; (void)out_size; (void)ws_size;
  const float* x     = (const float*)d_in[0];
  const float* pre   = (const float*)d_in[1];
  const float* mask  = (const float*)d_in[2];
  const float* q_w   = (const float*)d_in[3];
  const float* gamma = (const float*)d_in[4];
  const float* alpha = (const float*)d_in[5];
  const float* bn1_s = (const float*)d_in[6];
  const float* bn1_b = (const float*)d_in[7];
  const float* c1_w  = (const float*)d_in[8];
  const float* u1    = (const float*)d_in[9];
  const float* bn2_s = (const float*)d_in[11];
  const float* bn2_b = (const float*)d_in[12];
  const float* c2_w  = (const float*)d_in[13];
  const float* u2    = (const float*)d_in[14];
  const float* by_w  = (const float*)d_in[16];
  const float* u3    = (const float*)d_in[17];

  float* res  = (float*)d_out;
  float* attn = res + (size_t)BATCH * CH * NPIX;     // attention lives in d_out (fp32 output)

  char* wsp = (char*)d_ws;
  size_t off = 0;
  auto alloc = [&](size_t bytes) -> void* {
    void* p = wsp + off;
    off = (off + bytes + 255) & ~(size_t)255;
    return p;
  };
  unsigned short* attnb   = (unsigned short*)alloc((size_t)BATCH * NPIX * NPIX * 2);
  unsigned short* qT      = (unsigned short*)alloc((size_t)BATCH * NPIX * CQ * 2);
  unsigned short* xb_cn   = (unsigned short*)alloc((size_t)BATCH * CH * NPIX * 2);
  unsigned short* xbT     = (unsigned short*)alloc((size_t)BATCH * NPIX * CH * 2);
  unsigned short* preb_cn = (unsigned short*)alloc((size_t)BATCH * CH * NPIX * 2);
  float*          outT    = (float*)alloc((size_t)BATCH * NPIX * CH * 4);
  float*          cfT     = (float*)alloc((size_t)BATCH * NPIX * CH * 4);
  unsigned short* ybp     = (unsigned short*)alloc((size_t)BATCH * PW * PW * CH2 * 2);
  unsigned short* h0bp    = (unsigned short*)alloc((size_t)BATCH * PW * PW * CH2 * 2);
  float*          h1f     = (float*)alloc((size_t)BATCH * NPIX * CH * 4);
  unsigned short* h1bp    = (unsigned short*)alloc((size_t)BATCH * PW * PW * CH * 2);
  float*          h2f     = (float*)alloc((size_t)BATCH * NPIX * CH * 4);
  float*          scf     = (float*)alloc((size_t)BATCH * NPIX * CH * 4);
  unsigned short* wqb     = (unsigned short*)alloc((size_t)CQ * CH * 2);
  unsigned short* w1t     = (unsigned short*)alloc((size_t)9 * CH * CH2 * 2);
  unsigned short* w2t     = (unsigned short*)alloc((size_t)9 * CH * CH * 2);
  unsigned short* byt     = (unsigned short*)alloc((size_t)CH * CH2 * 2);
  float*          vtmp    = (float*)alloc(4608 * 4);
  float*          utmp    = (float*)alloc(256 * 4);
  float*          invs    = (float*)alloc(4 * 4);
  float*          a1 = (float*)alloc(CH2 * 4);
  float*          b1 = (float*)alloc(CH2 * 4);
  float*          a2 = (float*)alloc(CH * 4);
  float*          b2 = (float*)alloc(CH * 4);

  // ---- zero padded buffers (halo ring must be 0 for the implicit-GEMM convs) ----
  k_zero_u16<<<2048, TPB, 0, stream>>>(ybp,  (size_t)BATCH * PW * PW * CH2);
  k_zero_u16<<<2048, TPB, 0, stream>>>(h0bp, (size_t)BATCH * PW * PW * CH2);
  k_zero_u16<<<2048, TPB, 0, stream>>>(h1bp, (size_t)BATCH * PW * PW * CH);

  // ---- conversions ----
  k_convert<<<8192, TPB, 0, stream>>>(x, pre, xb_cn, xbT, preb_cn);
  k_convert_wq<<<64, TPB, 0, stream>>>(q_w, wqb);

  // ---- spectral norm + weight transforms ----
  k_sn<<<1, TPB, 0, stream>>>(c1_w, u1, 2 * CH * 9, vtmp, utmp, invs + 0);
  k_wt3x3<<<4608, TPB, 0, stream>>>(c1_w, w1t, invs + 0, CH2);
  k_sn<<<1, TPB, 0, stream>>>(c2_w, u2, CH * 9, vtmp, utmp, invs + 1);
  k_wt3x3<<<2304, TPB, 0, stream>>>(c2_w, w2t, invs + 1, CH);
  k_sn<<<1, TPB, 0, stream>>>(by_w, u3, 2 * CH, vtmp, utmp, invs + 2);
  k_wt1x1<<<512, TPB, 0, stream>>>(by_w, byt, invs + 2);

  // ---- attention ----
  k_qgemm<<<256, TPB, 0, stream>>>(xbT, wqb, qT);                    // 2048 wave-tiles
  k_energy<<<4096, TPB, 0, stream>>>(qT, attn);                      // 32768 wave tile-groups
  k_softmax<<<BATCH * NPIX, TPB, 0, stream>>>(attn, attnb);          // 8192 rows
  k_attn_apply<<<512, TPB, 0, stream>>>(attnb, xb_cn, preb_cn, outT, cfT);

  // ---- y / ResBlock ----
  k_build_y<<<16384, TPB, 0, stream>>>(outT, cfT, x, pre, mask, gamma, alpha, ybp);
  k_bnstats_pad<<<CH2, TPB, 0, stream>>>(ybp, CH2, bn1_s, bn1_b, a1, b1);
  k_act_pad<<<16384, TPB, 0, stream>>>(ybp, h0bp, a1, b1, 9);
  k_conv3x3<<<256, TPB, 0, stream>>>(h0bp, w1t, h1f, CH2);
  k_bnstats_f32<<<CH, TPB, 0, stream>>>(h1f, CH, bn2_s, bn2_b, a2, b2);
  k_act_f32<<<8192, TPB, 0, stream>>>(h1f, h1bp, a2, b2);
  k_conv3x3<<<256, TPB, 0, stream>>>(h1bp, w2t, h2f, CH);
  k_conv1x1<<<256, TPB, 0, stream>>>(ybp, byt, scf);
  k_final<<<8192, TPB, 0, stream>>>(h2f, scf, res);
}
